// Model_83330955477617
// MI455X (gfx1250) — compile-verified, matrix operand order checked by reading, not run on hardware
//
#include <hip/hip_runtime.h>
#include <hip/hip_bf16.h>

// ---------------- problem constants ----------------
#define NN   50000          // nodes
#define NE   800000         // edges (before self loops)
#define NEE  850000         // edges + self loops
#define BB   1024           // batch
#define DD   64             // hidden
#define HH   6              // heads
#define HD   384            // H*D

// WMMA vector types (probe-confirmed signatures)
typedef __attribute__((ext_vector_type(16))) __bf16 v16bf;
typedef __attribute__((ext_vector_type(8)))  float  v8f;
// TDM descriptor groups (clang-23 / therock 6-arg builtin form)
typedef __attribute__((ext_vector_type(4))) unsigned int u32x4;
typedef __attribute__((ext_vector_type(8))) int i32x8;
typedef __attribute__((ext_vector_type(4))) int i32x4;

// K index pattern for 16-bit A/B fragments (ISA 7.12.2):
// lanes 0-15: VGPR0..3 -> K 0..7, VGPR4..7 -> K 16..23
// lanes 16-31: VGPR0..3 -> K 8..15, VGPR4..7 -> K 24..31
__device__ __forceinline__ int kpat(int e, int half) {
  return (e < 8) ? (8 * half + e) : (16 + 8 * half + (e - 8));
}

// float atomic max via signed/unsigned int atomics (no CAS loop)
__device__ __forceinline__ void atomicMaxF(float* addr, float val) {
  if (val >= 0.f) atomicMax((int*)addr, __float_as_int(val));
  else            atomicMin((unsigned int*)addr, __float_as_uint(val));
}

__device__ __forceinline__ int rfl(unsigned v) {
  return __builtin_amdgcn_readfirstlane((int)v);
}

// Issue a TDM load of a (4 rows x 32 f32) sub-tile of `item` into LDS.
// cols_rem = NN - k0 so out-of-tensor reads in the final partial K-tile
// return zero (ISA 8.2: reads beyond tensor dimension return zero).
__device__ __forceinline__ void tdm_load_rows4(const float* gptr, unsigned lds_off,
                                               int cols_rem) {
  unsigned long long ga = (unsigned long long)(uintptr_t)gptr;
  unsigned td0 = (unsigned)cols_rem;
  u32x4 g0;
  g0[0] = (unsigned)rfl(1u);                                   // count=1 (valid, user)
  g0[1] = (unsigned)rfl(lds_off);                              // lds_addr (bytes)
  g0[2] = (unsigned)rfl((unsigned)ga);                         // global_addr[31:0]
  g0[3] = (unsigned)rfl(((unsigned)(ga >> 32) & 0x01ffffffu)   // global_addr[56:32]
                        | (2u << 30));                         // type=2 (image)
  i32x8 g1;
  g1[0] = rfl(2u << 16);                       // wg_mask=0, data_size=2 (4 bytes)
  g1[1] = rfl((td0 & 0xffffu) << 16);          // tensor_dim0[15:0] @ bit48
  g1[2] = rfl((td0 >> 16) | (4u << 16));       // tensor_dim0[31:16] | tensor_dim1=4
  g1[3] = rfl(32u << 16);                      // tensor_dim1 hi=0 | tile_dim0=32
  g1[4] = rfl(4u);                             // tile_dim1=4, tile_dim2=0
  g1[5] = rfl((unsigned)NN);                   // tensor_dim0_stride[31:0]
  g1[6] = rfl(0u);                             // stride hi / dim1_stride lo
  g1[7] = rfl(0u);
  i32x4 gz4;
  gz4[0] = rfl(0u); gz4[1] = rfl(0u); gz4[2] = rfl(0u); gz4[3] = rfl(0u);
  i32x8 gz8;
  gz8[0] = rfl(0u); gz8[1] = rfl(0u); gz8[2] = rfl(0u); gz8[3] = rfl(0u);
  gz8[4] = rfl(0u); gz8[5] = rfl(0u); gz8[6] = rfl(0u); gz8[7] = rfl(0u);
  __builtin_amdgcn_tensor_load_to_lds(g0, g1, gz4, gz4, gz8, 0);
}

// ---------------- init kernels ----------------
__global__ void k_init_layer(float* __restrict__ acc, float* __restrict__ segm,
                             float* __restrict__ segs) {
  size_t i = (size_t)blockIdx.x * blockDim.x + threadIdx.x;
  if (i < (size_t)NN * HD) acc[i] = 0.f;
  if (i < (size_t)NN * HH) { segs[i] = 0.f; segm[i] = __int_as_float(0xff800000); }
}

__global__ void k_init_pool(float* __restrict__ ip) {
  int i = blockIdx.x * blockDim.x + threadIdx.x;
  if (i < BB * DD) ip[i] = 0.f;
}

// ---------------- xp = x @ W   (50000x64 @ 64x384), one wave per 16x16 tile ----
__global__ void k_xp_gemm(const float* __restrict__ x, const float* __restrict__ W,
                          float* __restrict__ xp) {
  int lane = threadIdx.x & 31;
  int n0 = blockIdx.x * 16;          // 24 N-tiles
  int m0 = blockIdx.y * 16;          // 3125 M-tiles
  int r = lane & 15, half = lane >> 4;
  v8f c = {};
#pragma unroll
  for (int kb = 0; kb < 2; ++kb) {   // K = 64 = 2 x 32
    int k0 = kb * 32;
    v16bf a, b;
#pragma unroll
    for (int e = 0; e < 16; ++e) {
      int k = k0 + kpat(e, half);
      a[e] = (__bf16)x[(size_t)(m0 + r) * DD + k];     // A: row per lane
      b[e] = (__bf16)W[(size_t)k * HD + n0 + r];       // B: col per lane
    }
    c = __builtin_amdgcn_wmma_f32_16x16x32_bf16(false, a, false, b, (short)0, c,
                                                false, false);
  }
#pragma unroll
  for (int g = 0; g < 8; ++g)
    xp[(size_t)(m0 + g + 8 * half) * HD + n0 + r] = c[g];
}

// ---------------- per-node attention coefficients ----------------
__global__ void k_attn(const float* __restrict__ xp, const float* __restrict__ asrc,
                       const float* __restrict__ adst, float* __restrict__ ans,
                       float* __restrict__ andv) {
  int i = blockIdx.x * blockDim.x + threadIdx.x;
  if (i >= NN * HH) return;
  int n = i / HH, h = i % HH;
  const float* xr = xp + (size_t)n * HD + h * DD;
  float s0 = 0.f, s1 = 0.f;
#pragma unroll 8
  for (int d = 0; d < DD; ++d) {
    float v = xr[d];
    s0 += v * asrc[h * DD + d];
    s1 += v * adst[h * DD + d];
  }
  ans[i] = s0; andv[i] = s1;
}

// ---------------- edge pass 1: segment max over dst ----------------
__global__ void k_edge_max(const int* __restrict__ ei, const float* __restrict__ ans,
                           const float* __restrict__ andv, float* __restrict__ segm) {
  int e = blockIdx.x * blockDim.x + threadIdx.x;
  if (e >= NEE) return;
  int s, d;
  if (e < NE) { s = ei[e]; d = ei[NE + e]; } else { s = d = e - NE; }
#pragma unroll
  for (int h = 0; h < HH; ++h) {
    float v = ans[s * HH + h] + andv[d * HH + h];
    v = v > 0.f ? v : 0.2f * v;
    atomicMaxF(&segm[d * HH + h], v);
  }
}

// ---------------- edge pass 2: exp + segment sum ----------------
__global__ void k_edge_expsum(const int* __restrict__ ei, const float* __restrict__ ans,
                              const float* __restrict__ andv, const float* __restrict__ segm,
                              float* __restrict__ segs, float* __restrict__ pbuf) {
  int e = blockIdx.x * blockDim.x + threadIdx.x;
  if (e >= NEE) return;
  int s, d;
  if (e < NE) { s = ei[e]; d = ei[NE + e]; } else { s = d = e - NE; }
#pragma unroll
  for (int h = 0; h < HH; ++h) {
    float v = ans[s * HH + h] + andv[d * HH + h];
    v = v > 0.f ? v : 0.2f * v;
    float p = __expf(v - segm[d * HH + h]);
    pbuf[(size_t)e * HH + h] = p;
    atomicAdd(&segs[d * HH + h], p);
  }
}

// ---------------- edge pass 3: weighted scatter (one wave32 per edge) --------
__global__ void k_edge_scatter(const int* __restrict__ ei, const float* __restrict__ pbuf,
                               const float* __restrict__ segs, const float* __restrict__ xp,
                               float* __restrict__ acc) {
  int wave = blockIdx.x * (blockDim.x >> 5) + (threadIdx.x >> 5);
  if (wave >= NEE) return;
  int lane = threadIdx.x & 31;
  int e = wave, s, d;
  if (e < NE) { s = ei[e]; d = ei[NE + e]; } else { s = d = e - NE; }
  float alpha[HH];
#pragma unroll
  for (int h = 0; h < HH; ++h)
    alpha[h] = pbuf[(size_t)e * HH + h] / (segs[(size_t)d * HH + h] + 1e-16f);
  const float* xs = xp + (size_t)s * HD;
  float* ao = acc + (size_t)d * HD;
#pragma unroll
  for (int t = 0; t < 12; ++t) {           // 12*32 = 384 = H*D
    int idx = lane + t * 32;               // idx>>6 == t>>1 (constant per iter)
    atomicAdd(&ao[idx], alpha[t >> 1] * xs[idx]);
  }
}

// ---------------- head mean + bias ----------------
__global__ void k_finalize(const float* __restrict__ acc, const float* __restrict__ bias,
                           float* __restrict__ xo) {
  int i = blockIdx.x * blockDim.x + threadIdx.x;
  if (i >= NN * DD) return;
  int n = i >> 6, d = i & 63;
  float sm = 0.f;
#pragma unroll
  for (int h = 0; h < HH; ++h) sm += acc[(size_t)n * HD + h * DD + d];
  xo[i] = sm * (1.f / 6.f) + bias[d];
}

// ---------------- i_pool = item @ x1  (1024x50000 @ 50000x64) ----------------
// Split-K WMMA; A tile staged via TDM (tensor_load_to_lds) with double
// buffering: DMA of tile st+1 overlaps WMMA of tile st, paced by TENSORcnt.
#define KSTEPS_TOTAL 1563            // ceil(50000/32)
#define SLABS 24
#define STEPS_PER_SLAB 66            // 24*66 >= 1563
__global__ void k_pool_gemm(const float* __restrict__ item, const float* __restrict__ xn,
                            float* __restrict__ ip) {
  __shared__ float At[2][16][32];    // double-buffered 16x32 f32 A tile
  int tid = threadIdx.x;
  int lane = tid & 31, wave = tid >> 5;
  int r = lane & 15, half = lane >> 4;
  int m0 = blockIdx.x * 16;
  int step0 = blockIdx.y * STEPS_PER_SLAB;
  int step1 = min(step0 + STEPS_PER_SLAB, KSTEPS_TOTAL);
  int n0 = wave * 16;                // 4 waves cover N = 64
  int row0 = wave * 4;               // each wave DMAs its own 4-row quarter
  const float* abase = item + (size_t)(m0 + row0) * NN;
  // LDS byte offsets of this wave's quarters (generic ptr low 32 bits = LDS off)
  unsigned lds0 = (unsigned)(uintptr_t)&At[0][row0][0];
  unsigned lds1 = (unsigned)(uintptr_t)&At[1][row0][0];
  v8f c = {};
  {  // prologue: DMA first tile into buffer 'step0 & 1'
    int k0 = step0 * 32;
    tdm_load_rows4(abase + k0, (step0 & 1) ? lds1 : lds0, NN - k0);
  }
  for (int st = step0; st < step1; ++st) {
    int cur = st & 1;
    __builtin_amdgcn_s_wait_tensorcnt(0);      // this wave's quarter landed
    __syncthreads();                           // all quarters visible
    if (st + 1 < step1) {                      // overlap next DMA with compute
      int k1 = (st + 1) * 32;
      tdm_load_rows4(abase + k1, cur ? lds0 : lds1, NN - k1);
    }
    int k0 = st * 32;
    v16bf a, b;
#pragma unroll
    for (int e = 0; e < 16; ++e) {
      int kk = kpat(e, half);
      a[e] = (__bf16)At[cur][r][kk];
      int k = k0 + kk;
      b[e] = (k < NN) ? (__bf16)xn[(size_t)k * DD + n0 + r] : (__bf16)0.f;
    }
    c = __builtin_amdgcn_wmma_f32_16x16x32_bf16(false, a, false, b, (short)0, c,
                                                false, false);
    __syncthreads();                           // buffer reuse fence (st+2 issue)
  }
#pragma unroll
  for (int g = 0; g < 8; ++g)
    atomicAdd(&ip[(size_t)(m0 + g + 8 * half) * DD + n0 + r], c[g]);
}

// ---------------- MLP head: concat -> 128 -> 1 -> sigmoid ----------------
__global__ void k_head(const int* __restrict__ user, const float* __restrict__ utab,
                       const float* __restrict__ ipool, const float* __restrict__ W0,
                       const float* __restrict__ b0, const float* __restrict__ W1,
                       const float* __restrict__ b1, float* __restrict__ out) {
  int b = blockIdx.x * blockDim.x + threadIdx.x;
  if (b >= BB) return;
  float in[2 * DD];
  const float* ur = utab + (size_t)user[b] * DD;
  for (int k = 0; k < DD; ++k) {
    in[k]      = ur[k];
    in[DD + k] = ipool[(size_t)b * DD + k];
  }
  float a1 = b1[0];
  for (int j = 0; j < 128; ++j) {
    float hj = b0[j];
    for (int k = 0; k < 128; ++k) hj += in[k] * W0[k * 128 + j];
    a1 += hj * W1[j];
  }
  out[b] = 1.f / (1.f + __expf(-a1));
}

// ---------------- launcher ----------------
extern "C" void kernel_launch(void* const* d_in, const int* in_sizes, int n_in,
                              void* d_out, int out_size, void* d_ws, size_t ws_size,
                              hipStream_t stream) {
  (void)in_sizes; (void)n_in; (void)out_size; (void)ws_size;
  const int*   user = (const int*)  d_in[0];
  const float* item = (const float*)d_in[1];
  const int*   ei   = (const int*)  d_in[3];
  const float* utab = (const float*)d_in[4];
  const float* itab = (const float*)d_in[5];
  const float* W[2]    = {(const float*)d_in[6],  (const float*)d_in[10]};
  const float* asrc[2] = {(const float*)d_in[7],  (const float*)d_in[11]};
  const float* adst[2] = {(const float*)d_in[8],  (const float*)d_in[12]};
  const float* gb[2]   = {(const float*)d_in[9],  (const float*)d_in[13]};
  const float* lW0 = (const float*)d_in[14];
  const float* lb0 = (const float*)d_in[15];
  const float* lW1 = (const float*)d_in[16];
  const float* lb1 = (const float*)d_in[17];
  float* out = (float*)d_out;

  float* ws = (float*)d_ws;
  size_t off = 0;
  float* xp    = ws + off; off += (size_t)NN * HD;
  float* acc   = ws + off; off += (size_t)NN * HD;
  float* x0    = ws + off; off += (size_t)NN * DD;
  float* x1    = ws + off; off += (size_t)NN * DD;
  float* ans   = ws + off; off += (size_t)NN * HH;
  float* andv  = ws + off; off += (size_t)NN * HH;
  float* segm  = ws + off; off += (size_t)NN * HH;
  float* segs  = ws + off; off += (size_t)NN * HH;
  float* pbuf  = ws + off; off += (size_t)NEE * HH;
  float* ipool = ws + off; off += (size_t)BB * DD;

  const float* xin = itab;
  float* xouts[2] = {x0, x1};
  for (int l = 0; l < 2; ++l) {
    k_init_layer<<<(NN * HD + 255) / 256, 256, 0, stream>>>(acc, segm, segs);
    k_xp_gemm<<<dim3(HD / 16, NN / 16), 32, 0, stream>>>(xin, W[l], xp);
    k_attn<<<(NN * HH + 255) / 256, 256, 0, stream>>>(xp, asrc[l], adst[l], ans, andv);
    k_edge_max<<<(NEE + 255) / 256, 256, 0, stream>>>(ei, ans, andv, segm);
    k_edge_expsum<<<(NEE + 255) / 256, 256, 0, stream>>>(ei, ans, andv, segm, segs, pbuf);
    k_edge_scatter<<<(NEE + 7) / 8, 256, 0, stream>>>(ei, pbuf, segs, xp, acc);
    k_finalize<<<(NN * DD + 255) / 256, 256, 0, stream>>>(acc, gb[l], xouts[l]);
    xin = xouts[l];
  }
  k_init_pool<<<(BB * DD + 255) / 256, 256, 0, stream>>>(ipool);
  k_pool_gemm<<<dim3(BB / 16, SLABS), 128, 0, stream>>>(item, x1, ipool);
  k_head<<<(BB + 255) / 256, 256, 0, stream>>>(user, utab, ipool, lW0, lb0, lW1, lb1, out);
}
